// Encoding_layer_50843822850433
// MI455X (gfx1250) — matrix operand order repaired and checked by reading, not compiled, not run on hardware
//
#include <hip/hip_runtime.h>

#define B_  16
#define L_  1024
#define U_  512
#define BL_ (B_*L_)

typedef __attribute__((ext_vector_type(16))) _Float16 v16h;
typedef __attribute__((ext_vector_type(8)))  _Float16 v8h;
typedef __attribute__((ext_vector_type(8)))  float    v8f;

__device__ __forceinline__ float sigmoidf_(float x) { return 1.0f / (1.0f + __expf(-x)); }

// ---------------- CDNA5 async global->LDS (ASYNCcnt path) ----------------
__device__ __forceinline__ void async_b128(const _Float16* g, _Float16* lds) {
  unsigned l = (unsigned)(uintptr_t)lds;   // low 32 bits of generic ptr = LDS offset
  asm volatile("global_load_async_to_lds_b128 %0, %1, off"
               :: "v"(l), "v"(g) : "memory");
}
__device__ __forceinline__ void wait_async0() {
  asm volatile("s_wait_asynccnt 0" ::: "memory");
}

// ---------------- fragment loaders (all vector loads) ----------------
// A fragment (16x32 f16) from a ROW-MAJOR f16 buffer, tile origin = base, leading dim ld.
__device__ __forceinline__ v16h frag_a_rm(const _Float16* base, int ld) {
  const int lane = threadIdx.x & 31;
  const _Float16* p = base + (size_t)(lane & 15) * ld + ((lane >> 4) << 3);
  v8h lo = *(const v8h*)p;
  v8h hi = *(const v8h*)(p + 16);
  return __builtin_shufflevector(lo, hi, 0,1,2,3,4,5,6,7,8,9,10,11,12,13,14,15);
}

// B fragment (32x16 f16) from an N-MAJOR f16 buffer: element(K=k,N=n) at base[n*ld + k].
__device__ __forceinline__ v16h frag_b_nm(const _Float16* base, int ld) {
  const int lane = threadIdx.x & 31;
  const _Float16* p = base + (size_t)(lane & 15) * ld + ((lane >> 4) << 3);
  v8h lo = *(const v8h*)p;
  v8h hi = *(const v8h*)(p + 16);
  return __builtin_shufflevector(lo, hi, 0,1,2,3,4,5,6,7,8,9,10,11,12,13,14,15);
}

#define WMMA_F16(A,Bb,C) __builtin_amdgcn_wmma_f32_16x16x32_f16(false, (A), false, (Bb), (short)0, (C), false, false)

// ---------------- prep: f32 -> f16 ----------------
__global__ __launch_bounds__(256)
void cvt_kernel(const float* __restrict__ x, _Float16* __restrict__ y) {
  size_t i = ((size_t)blockIdx.x * 256 + threadIdx.x) * 4;
  float4 v = *(const float4*)(x + i);
  y[i+0] = (_Float16)v.x; y[i+1] = (_Float16)v.y;
  y[i+2] = (_Float16)v.z; y[i+3] = (_Float16)v.w;
}

// ---------------- prep: W (KxN f32) -> WT (NxK f16) ----------------
__global__ __launch_bounds__(256)
void wT_kernel(const float* __restrict__ W, _Float16* __restrict__ WT, int K, int N) {
  __shared__ float tile[32][33];
  const int k0 = blockIdx.y * 32, n0 = blockIdx.x * 32;
  const int r = threadIdx.x >> 5, c = threadIdx.x & 31;
#pragma unroll
  for (int i = 0; i < 4; ++i)
    tile[r + i*8][c] = W[(size_t)(k0 + r + i*8) * N + n0 + c];
  __syncthreads();
#pragma unroll
  for (int i = 0; i < 4; ++i)
    WT[(size_t)(n0 + r + i*8) * K + k0 + c] = (_Float16)tile[c][r + i*8];
}

// ---------------- prep: X (B,L,U f16) -> XT (B,U,L f16) ----------------
__global__ __launch_bounds__(256)
void xT_kernel(const _Float16* __restrict__ X, _Float16* __restrict__ XT) {
  __shared__ float tile[32][33];
  const int b = blockIdx.z, j0 = blockIdx.y * 32, d0 = blockIdx.x * 32;
  const int r = threadIdx.x >> 5, c = threadIdx.x & 31;
#pragma unroll
  for (int i = 0; i < 4; ++i)
    tile[r + i*8][c] = (float)X[((size_t)b * L_ + j0 + r + i*8) * U_ + d0 + c];
  __syncthreads();
#pragma unroll
  for (int i = 0; i < 4; ++i)
    XT[((size_t)b * U_ + d0 + r + i*8) * L_ + j0 + c] = (_Float16)tile[c][r + i*8];
}

// ========== highway: x' = relu(x@tW+tb)*sig(x@cW+cb) + x*(1-sig) ==========
// 64x32 block tile, double-buffered LDS filled by async-to-LDS b128.
__global__ __launch_bounds__(256)
void highway_kernel(const _Float16* __restrict__ xin,
                    const _Float16* __restrict__ tWT, const float* __restrict__ tb,
                    const _Float16* __restrict__ cWT, const float* __restrict__ cb,
                    _Float16* __restrict__ xout) {
  __shared__ _Float16 As[2][64 * 32];
  __shared__ _Float16 Bt[2][32 * 32];
  __shared__ _Float16 Bc[2][32 * 32];
  const int tid = threadIdx.x, wave = tid >> 5;
  const int row0 = blockIdx.y * 64, col0 = blockIdx.x * 32;

  // this thread's staging slot (one 16B chunk of A, one of Bt or Bc)
  const int ar = tid >> 2, ac8 = (tid & 3) << 3;
  const int bm = tid >> 7, bidx = tid & 127;
  const int bn = bidx >> 2, bk8 = (bidx & 3) << 3;
  const _Float16* aG = xin + (size_t)(row0 + ar) * U_ + ac8;
  const _Float16* bG = (bm ? cWT : tWT) + (size_t)(col0 + bn) * U_ + bk8;

  v8f acct = {}, accc = {};
#define HW_ISSUE(buf, k0)                                         \
  { async_b128(aG + (k0), &As[buf][ar * 32 + ac8]);               \
    async_b128(bG + (k0), bm ? &Bc[buf][bn * 32 + bk8]            \
                             : &Bt[buf][bn * 32 + bk8]); }
  HW_ISSUE(0, 0)
  for (int k0 = 0, i = 0; k0 < U_; k0 += 32, ++i) {
    const int cur = i & 1;
    wait_async0();
    __syncthreads();                       // tile `cur` complete in LDS for all waves
    if (k0 + 32 < U_) HW_ISSUE(cur ^ 1, k0 + 32)
    v16h a  = frag_a_rm(&As[cur][(wave >> 1) * 16 * 32], 32);
    v16h bt = frag_b_nm(&Bt[cur][(wave & 1) * 16 * 32], 32);
    v16h bc = frag_b_nm(&Bc[cur][(wave & 1) * 16 * 32], 32);
    acct = WMMA_F16(a, bt, acct);
    accc = WMMA_F16(a, bc, accc);
  }
#undef HW_ISSUE

  const int lane = tid & 31, n = lane & 15, mb = (lane >> 4) << 3;
  const int gcol = col0 + (wave & 1) * 16 + n;
#pragma unroll
  for (int r = 0; r < 8; ++r) {
    int grow = row0 + (wave >> 1) * 16 + mb + r;
    float t = acct[r] + tb[gcol]; t = t > 0.f ? t : 0.f;
    float c = sigmoidf_(accc[r] + cb[gcol]);
    float xv = (float)xin[(size_t)grow * U_ + gcol];
    xout[(size_t)grow * U_ + gcol] = (_Float16)(t * c + xv * (1.f - c));
  }
}

// ========== s1/s2 per-row dots ==========
__global__ __launch_bounds__(256)
void s12_kernel(const _Float16* __restrict__ x, const float* __restrict__ aW,
                float* __restrict__ s1, float* __restrict__ s2) {
  const int row = blockIdx.x * 8 + (threadIdx.x >> 5);
  const int lane = threadIdx.x & 31;
  const _Float16* xr = x + (size_t)row * U_;
  float a1 = 0.f, a2 = 0.f;
  for (int d = lane; d < U_; d += 32) {
    float xv = (float)xr[d];
    a1 += xv * aW[d];
    a2 += xv * aW[U_ + d];
  }
#pragma unroll
  for (int m = 16; m >= 1; m >>= 1) {
    a1 += __shfl_xor(a1, m, 32);
    a2 += __shfl_xor(a2, m, 32);
  }
  if (lane == 0) { s1[row] = a1; s2[row] = a2; }
}

// ========== flash-style additive attention, BM=32 query rows, BN=128 keys ==========
__global__ __launch_bounds__(256)
void attn_kernel(const _Float16* __restrict__ xh, const _Float16* __restrict__ xT,
                 const float* __restrict__ s1g, const float* __restrict__ s2g,
                 const float* __restrict__ w3, const float* __restrict__ abp,
                 _Float16* __restrict__ atth) {
  __shared__ _Float16 qs[32 * 512];   // 32 KB: Q rows pre-scaled by w3
  __shared__ _Float16 ps[32 * 128];   // 8 KB : P tile
  __shared__ float s2t[128], srow1[32], red[4 * 32];
  __shared__ float rowm[32], rowsc[32], rowl[32];

  const int tid = threadIdx.x, wave = tid >> 5, lane = tid & 31;
  const int mrow = wave >> 2, cg = wave & 3;
  const int b = blockIdx.y, i0 = blockIdx.x * 32;
  const float ab = abp[0];
  const int n = lane & 15, mb = (lane >> 4) << 3;
  const int rowi = mrow * 16 + mb;

  for (int idx = tid; idx < 32 * 512; idx += 256) {
    int r = idx >> 9, d = idx & 511;
    qs[idx] = (_Float16)((float)xh[((size_t)b * L_ + i0 + r) * U_ + d] * w3[d]);
  }
  if (tid < 32) { srow1[tid] = s1g[b * L_ + i0 + tid]; rowm[tid] = -1e30f; rowl[tid] = 0.f; }
  v8f o[8] = {};
  __syncthreads();

  for (int j0 = 0; j0 < L_; j0 += 128) {
    if (tid < 128) s2t[tid] = s2g[b * L_ + j0 + tid];
    __syncthreads();

    // ---- S = Q @ X^T : two 16x16 tiles per wave (cols cg*32 .. cg*32+31)
    v8f s0 = {}, s1v = {};
    const _Float16* qbase = qs + mrow * 16 * 512;
    const _Float16* xrow0 = xh + ((size_t)b * L_ + j0 + cg * 32) * U_;
    for (int kk = 0; kk < U_; kk += 32) {
      v16h a  = frag_a_rm(qbase + kk, 512);
      v16h b0 = frag_b_nm(xrow0 + kk, 512);
      v16h b1 = frag_b_nm(xrow0 + (size_t)16 * 512 + kk, 512);
      s0  = WMMA_F16(a, b0, s0);
      s1v = WMMA_F16(a, b1, s1v);
    }
    { // bias + relu
      float s2a = s2t[cg * 32 + n], s2b = s2t[cg * 32 + 16 + n];
#pragma unroll
      for (int r = 0; r < 8; ++r) {
        float v0 = s0[r]  + srow1[rowi + r] + s2a + ab;
        float v1 = s1v[r] + srow1[rowi + r] + s2b + ab;
        s0[r]  = v0 > 0.f ? v0 : 0.f;
        s1v[r] = v1 > 0.f ? v1 : 0.f;
      }
    }
    // ---- partial row max over this wave's 32 cols
    float pm[8];
#pragma unroll
    for (int r = 0; r < 8; ++r) pm[r] = fmaxf(s0[r], s1v[r]);
#pragma unroll
    for (int m = 8; m >= 1; m >>= 1)
#pragma unroll
      for (int r = 0; r < 8; ++r) pm[r] = fmaxf(pm[r], __shfl_xor(pm[r], m, 32));
    if (n == 0)
#pragma unroll
      for (int r = 0; r < 8; ++r) red[cg * 32 + rowi + r] = pm[r];
    __syncthreads();
    if (tid < 32) {
      float mx = fmaxf(fmaxf(red[tid], red[32 + tid]), fmaxf(red[64 + tid], red[96 + tid]));
      float mold = rowm[tid], mnew = fmaxf(mold, mx);
      float sc = __expf(mold - mnew);
      rowm[tid] = mnew; rowsc[tid] = sc; rowl[tid] *= sc;
    }
    __syncthreads();
    // ---- P = exp(S - m), store f16, row sums
    float psum[8];
#pragma unroll
    for (int r = 0; r < 8; ++r) {
      float p0 = __expf(s0[r]  - rowm[rowi + r]);
      float p1 = __expf(s1v[r] - rowm[rowi + r]);
      psum[r] = p0 + p1;
      ps[(size_t)(rowi + r) * 128 + cg * 32 + n]      = (_Float16)p0;
      ps[(size_t)(rowi + r) * 128 + cg * 32 + 16 + n] = (_Float16)p1;
    }
#pragma unroll
    for (int m = 8; m >= 1; m >>= 1)
#pragma unroll
      for (int r = 0; r < 8; ++r) psum[r] += __shfl_xor(psum[r], m, 32);
    if (n == 0)
#pragma unroll
      for (int r = 0; r < 8; ++r) red[cg * 32 + rowi + r] = psum[r];
    __syncthreads();
    if (tid < 32) rowl[tid] += red[tid] + red[32 + tid] + red[64 + tid] + red[96 + tid];
    // ---- rescale O, then O += P @ X (K = 128), B from d-major xT
#pragma unroll
    for (int t = 0; t < 8; ++t)
#pragma unroll
      for (int r = 0; r < 8; ++r) o[t][r] *= rowsc[rowi + r];
    for (int kk = 0; kk < 128; kk += 32) {
      v16h a = frag_a_rm(ps + mrow * 16 * 128 + kk, 128);
#pragma unroll
      for (int t = 0; t < 8; ++t) {
        int dcol = cg * 128 + t * 16;
        v16h bb = frag_b_nm(xT + ((size_t)b * U_ + dcol) * L_ + j0 + kk, L_);
        o[t] = WMMA_F16(a, bb, o[t]);
      }
    }
    __syncthreads();
  }

#pragma unroll
  for (int t = 0; t < 8; ++t) {
    int dcol = cg * 128 + t * 16 + n;
#pragma unroll
    for (int r = 0; r < 8; ++r) {
      int grow = i0 + rowi + r;
      atth[((size_t)b * L_ + grow) * U_ + dcol] = (_Float16)(o[t][r] / rowl[rowi + r]);
    }
  }
}

// ========== output: z=sig(concat@ffW+ffb), r=sig(concat@frW+frb); out=r*in+z*z ==========
__global__ __launch_bounds__(256)
void out_kernel(const float* __restrict__ inpf,
                const _Float16* __restrict__ inh, const _Float16* __restrict__ atth,
                const _Float16* __restrict__ ffWT, const _Float16* __restrict__ frWT,
                const float* __restrict__ ffb, const float* __restrict__ frb,
                float* __restrict__ out) {
  __shared__ _Float16 As[2][64 * 32];
  __shared__ _Float16 Bf[2][32 * 32];
  __shared__ _Float16 Br[2][32 * 32];
  const int tid = threadIdx.x, wave = tid >> 5;
  const int row0 = blockIdx.y * 64, col0 = blockIdx.x * 32;

  const int ar = tid >> 2, ac8 = (tid & 3) << 3;
  const int bm = tid >> 7, bidx = tid & 127;
  const int bn = bidx >> 2, bk8 = (bidx & 3) << 3;
  const _Float16* aG0 = inh  + (size_t)(row0 + ar) * U_ + ac8;
  const _Float16* aG1 = atth + (size_t)(row0 + ar) * U_ + ac8;
  const _Float16* bG  = (bm ? frWT : ffWT) + (size_t)(col0 + bn) * (2 * U_) + bk8;

  v8f accz = {}, accr = {};
#define OUT_ISSUE(buf, k0)                                                   \
  { const _Float16* ag = ((k0) < U_) ? (aG0 + (k0)) : (aG1 + ((k0) - U_));   \
    async_b128(ag, &As[buf][ar * 32 + ac8]);                                 \
    async_b128(bG + (k0), bm ? &Br[buf][bn * 32 + bk8]                       \
                             : &Bf[buf][bn * 32 + bk8]); }
  OUT_ISSUE(0, 0)
  for (int k0 = 0, i = 0; k0 < 2 * U_; k0 += 32, ++i) {
    const int cur = i & 1;
    wait_async0();
    __syncthreads();
    if (k0 + 32 < 2 * U_) OUT_ISSUE(cur ^ 1, k0 + 32)
    v16h a  = frag_a_rm(&As[cur][(wave >> 1) * 16 * 32], 32);
    v16h bz = frag_b_nm(&Bf[cur][(wave & 1) * 16 * 32], 32);
    v16h br = frag_b_nm(&Br[cur][(wave & 1) * 16 * 32], 32);
    accz = WMMA_F16(a, bz, accz);
    accr = WMMA_F16(a, br, accr);
  }
#undef OUT_ISSUE

  const int lane = tid & 31, n = lane & 15, mb = (lane >> 4) << 3;
  const int gcol = col0 + (wave & 1) * 16 + n;
#pragma unroll
  for (int r = 0; r < 8; ++r) {
    int grow = row0 + (wave >> 1) * 16 + mb + r;
    float z  = sigmoidf_(accz[r] + ffb[gcol]);
    float rr = sigmoidf_(accr[r] + frb[gcol]);
    float iv = inpf[(size_t)grow * U_ + gcol];
    out[(size_t)grow * U_ + gcol] = rr * iv + z * z;
  }
}

extern "C" void kernel_launch(void* const* d_in, const int* in_sizes, int n_in,
                              void* d_out, int out_size, void* d_ws, size_t ws_size,
                              hipStream_t stream) {
  (void)in_sizes; (void)n_in; (void)out_size; (void)ws_size;
  const float* inputs = (const float*)d_in[0];
  const float* tW  = (const float*)d_in[1];
  const float* tb  = (const float*)d_in[2];
  const float* cW  = (const float*)d_in[3];
  const float* cb  = (const float*)d_in[4];
  const float* aW  = (const float*)d_in[5];
  const float* ab  = (const float*)d_in[6];
  const float* frW = (const float*)d_in[7];
  const float* frb = (const float*)d_in[8];
  const float* ffW = (const float*)d_in[9];
  const float* ffb = (const float*)d_in[10];
  float* out = (float*)d_out;

  const size_t ACT = (size_t)BL_ * U_;
  const size_t WTS = (size_t)U_ * U_;
  char* w = (char*)d_ws;
  _Float16* inh  = (_Float16*)w;           w += ACT * 2;
  _Float16* x0h  = (_Float16*)w;           w += ACT * 2;
  _Float16* x1h  = (_Float16*)w;           w += ACT * 2;
  _Float16* x1T  = (_Float16*)w;           w += ACT * 2;
  _Float16* atth = (_Float16*)w;           w += ACT * 2;
  _Float16* tWT  = (_Float16*)w;           w += 2 * WTS * 2;
  _Float16* cWT  = (_Float16*)w;           w += 2 * WTS * 2;
  _Float16* ffWT = (_Float16*)w;           w += 2 * WTS * 2;
  _Float16* frWT = (_Float16*)w;           w += 2 * WTS * 2;
  float* s1b = (float*)w;                  w += (size_t)BL_ * 4;
  float* s2b = (float*)w;

  dim3 blk(256);
  cvt_kernel<<<(unsigned)(ACT / 1024), blk, 0, stream>>>(inputs, inh);
  wT_kernel<<<dim3(16, 16), blk, 0, stream>>>(tW,       tWT,       U_, U_);
  wT_kernel<<<dim3(16, 16), blk, 0, stream>>>(tW + WTS, tWT + WTS, U_, U_);
  wT_kernel<<<dim3(16, 16), blk, 0, stream>>>(cW,       cWT,       U_, U_);
  wT_kernel<<<dim3(16, 16), blk, 0, stream>>>(cW + WTS, cWT + WTS, U_, U_);
  wT_kernel<<<dim3(16, 32), blk, 0, stream>>>(ffW, ffWT, 2 * U_, U_);
  wT_kernel<<<dim3(16, 32), blk, 0, stream>>>(frW, frWT, 2 * U_, U_);

  dim3 ghw(U_ / 32, BL_ / 64);
  highway_kernel<<<ghw, blk, 0, stream>>>(inh, tWT, tb, cWT, cb, x0h);
  highway_kernel<<<ghw, blk, 0, stream>>>(x0h, tWT + WTS, tb + U_, cWT + WTS, cb + U_, x1h);

  s12_kernel<<<BL_ / 8, blk, 0, stream>>>(x1h, aW, s1b, s2b);
  xT_kernel<<<dim3(U_ / 32, L_ / 32, B_), blk, 0, stream>>>(x1h, x1T);

  attn_kernel<<<dim3(L_ / 32, B_), blk, 0, stream>>>(x1h, x1T, s1b, s2b,
                                                     aW + 2 * U_, ab, atth);
  out_kernel<<<ghw, blk, 0, stream>>>(inputs, inh, atth, ffWT, frWT, ffb, frb, out);
}
